// Seq2SeqRNN_11115375362488
// MI455X (gfx1250) — matrix operand order repaired and checked by reading, not compiled, not run on hardware
//
#include <hip/hip_runtime.h>
#include <hip/hip_bf16.h>

// ---------------------------------------------------------------------------
// Seq2Seq RNN for MI455X (gfx1250), wave32 + WMMA bf16 (fp32 accumulate).
// B=32, S=T=64, V=32000, E=512, H=1024.
// ---------------------------------------------------------------------------

typedef __attribute__((ext_vector_type(16))) __bf16 v16bf;
typedef __attribute__((ext_vector_type(8)))  float  v8f;

#define BB   32
#define SEQ  64
#define VV   32000
#define EE   512
#define HH   1024
#define MROWS (BB * SEQ)        // 2048

// ---- WMMA fragment helpers -------------------------------------------------
// A-matrix (16x32 bf16) lane layout: lane = m + 16*hi.
// VGPR i (elements 2i,2i+1): k = (i<4 ? 2i : 16+2(i-4)) + 8*hi (+0/+1).
__device__ __forceinline__ v16bf load_a_frag_f32(const float* __restrict__ row,
                                                 int kk, int hi) {
  v16bf a;
#pragma unroll
  for (int i = 0; i < 8; ++i) {
    int kb = ((i < 4) ? (2 * i) : (16 + 2 * (i - 4))) + 8 * hi + kk;
    float2 f = *(const float2*)(row + kb);
    a[2 * i]     = (__bf16)f.x;
    a[2 * i + 1] = (__bf16)f.y;
  }
  return a;
}

// B fragments are pre-swizzled: tile (kt,nt) stored as 32 lanes x 16 bf16,
// lane L holds column n = nt*16 + (L&15), K = kt*32 + (e<8 ? 8*(L>>4)+e : 16+8*(L>>4)+e-8).
__device__ __forceinline__ v16bf load_b_frag(const __bf16* __restrict__ wsw,
                                             int tile, int lane) {
  return *(const v16bf*)(wsw + ((size_t)tile * 32 + lane) * 16);
}

__device__ __forceinline__ v8f wmma_bf16(v16bf a, v16bf b, v8f c) {
  return __builtin_amdgcn_wmma_f32_16x16x32_bf16(false, a, false, b,
                                                 (short)0, c, false, false);
}

// ---- Weight prep: fp32 row-major [K,N] -> bf16 swizzled B fragments --------
__global__ void __launch_bounds__(256)
prep_bsw(const float* __restrict__ W, __bf16* __restrict__ out, int K, int N) {
  int tid = blockIdx.x * blockDim.x + threadIdx.x;
  int NT = N >> 4;
  int total = (K >> 5) * NT * 32;
  if (tid >= total) return;
  int lane = tid & 31;
  int tile = tid >> 5;
  int nt = tile % NT;
  int kt = tile / NT;
  int n  = nt * 16 + (lane & 15);
  int hi = lane >> 4;
  v16bf frag;
#pragma unroll
  for (int e = 0; e < 16; ++e) {
    int k = kt * 32 + ((e < 8) ? (8 * hi + e) : (16 + 8 * hi + (e - 8)));
    frag[e] = (__bf16)W[(size_t)k * N + n];
  }
  *(v16bf*)(out + (size_t)tid * 16) = frag;
}

// ---- Pack fp32 row-major [M,K] activations into bf16 A fragments -----------
// Fragment (mt, kt, lane): 16 contiguous bf16 at ((mt*KT + kt)*32 + lane)*16.
__global__ void __launch_bounds__(256)
pack_a_bf16(const float* __restrict__ src, __bf16* __restrict__ dst,
            int MT, int KT, int ldk) {
  int tid = blockIdx.x * blockDim.x + threadIdx.x;
  int total = MT * KT * 32;
  if (tid >= total) return;
  int lane = tid & 31;
  int tile = tid >> 5;
  int kt = tile % KT;
  int mt = tile / KT;
  int ml = lane & 15, hi = lane >> 4;
  const float* row = src + (size_t)(mt * 16 + ml) * ldk;
  v16bf a = load_a_frag_f32(row, kt * 32, hi);
  *(v16bf*)(dst + (size_t)tid * 16) = a;
}

// ---- x = emb[tok] @ Wih + bih  ([2048,512]@[512,1024]) ---------------------
__global__ void __launch_bounds__(256)
xproj(const int* __restrict__ tok, const float* __restrict__ emb,
      const __bf16* __restrict__ wsw, const float* __restrict__ bias,
      float* __restrict__ xout) {
  const int NT = HH / 16;                       // 64
  int wave = blockIdx.x * 8 + (threadIdx.x >> 5);
  int lane = threadIdx.x & 31;
  int mt = wave >> 6;
  int nt = wave & 63;
  int ml = lane & 15, hi = lane >> 4;
  int row = mt * 16 + ml;
  const float* arow = emb + (size_t)tok[row] * EE;
  v8f acc = {};
#pragma unroll 4
  for (int kt = 0; kt < EE / 32; ++kt) {
    v16bf a = load_a_frag_f32(arow, kt * 32, hi);
    v16bf b = load_b_frag(wsw, kt * NT + nt, lane);
    acc = wmma_bf16(a, b, acc);
  }
  int ncol = nt * 16 + ml;
  float bb = bias[ncol];
#pragma unroll
  for (int r = 0; r < 8; ++r) {
    int m = mt * 16 + r + 8 * hi;
    xout[(size_t)m * HH + ncol] = acc[r] + bb;
  }
}

// ---- one recurrent step: h_out = tanh(x[:,t,:] + h_in@Whh + bhh) -----------
__global__ void __launch_bounds__(256)
rnn_step(const float* __restrict__ h_in, const float* __restrict__ xbase,
         const __bf16* __restrict__ whh_sw, const float* __restrict__ bhh,
         float* __restrict__ h_out, float* __restrict__ hs, int t) {
  const int NT = HH / 16;                       // 64
  int wave = blockIdx.x * 8 + (threadIdx.x >> 5);   // 0..127
  int lane = threadIdx.x & 31;
  int mt = wave >> 6;                           // 0..1
  int nt = wave & 63;
  int ml = lane & 15, hi = lane >> 4;
  const float* arow = h_in + (size_t)(mt * 16 + ml) * HH;
  v8f acc = {};
#pragma unroll 4
  for (int kt = 0; kt < HH / 32; ++kt) {
    v16bf a = load_a_frag_f32(arow, kt * 32, hi);
    v16bf b = load_b_frag(whh_sw, kt * NT + nt, lane);
    acc = wmma_bf16(a, b, acc);
  }
  int ncol = nt * 16 + ml;
  float bb = bhh[ncol];
#pragma unroll
  for (int r = 0; r < 8; ++r) {
    int m = mt * 16 + r + 8 * hi;               // batch index 0..31
    float v = tanhf(acc[r] + xbase[((size_t)m * SEQ + t) * HH + ncol] + bb);
    h_out[(size_t)m * HH + ncol] = v;
    if (hs) hs[((size_t)t * BB + m) * HH + ncol] = v;
  }
}

// ---- logits = hs @ Wout + bout  ([2048,1024]@[1024,32000]) -----------------
// A is pre-packed bf16 fragments; wave computes a 16x80 slab (5 N-tiles).
// Blocks are mt-fast so the 65MB bf16 Wout panel streams from HBM once.
__global__ void __launch_bounds__(256)
out_gemm(const __bf16* __restrict__ hsp, const __bf16* __restrict__ wsw,
         const float* __restrict__ bout, float* __restrict__ out) {
  const int NT = VV / 16;                       // 2000
  const int KT = HH / 32;                       // 32
  int w = threadIdx.x >> 5, lane = threadIdx.x & 31;
  int mt  = blockIdx.x & 127;                   // 0..127
  int ntb = blockIdx.x >> 7;                    // 0..49
  int nt0 = ntb * 40 + w * 5;
  int ml = lane & 15, hi = lane >> 4;
  const v16bf* afrag = (const v16bf*)(hsp + ((size_t)mt * KT * 32 + lane) * 16);
  v8f acc[5] = {};
  for (int kt = 0; kt < KT; ++kt) {
    v16bf a = afrag[(size_t)kt * 32];
    // prefetch next K-tile of the streamed B panel
    if (kt + 1 < KT)
      __builtin_prefetch(wsw + (((size_t)(kt + 1) * NT + nt0) * 32 + lane) * 16, 0, 1);
#pragma unroll
    for (int j = 0; j < 5; ++j) {
      v16bf b = load_b_frag(wsw, kt * NT + nt0 + j, lane);
      acc[j] = wmma_bf16(a, b, acc[j]);
    }
  }
#pragma unroll
  for (int j = 0; j < 5; ++j) {
    int ncol = (nt0 + j) * 16 + ml;
    float bb = bout[ncol];
#pragma unroll
    for (int r = 0; r < 8; ++r) {
      int m = mt * 16 + r + 8 * hi;             // m = t*B + b
      int tt = m >> 5, bidx = m & 31;
      out[((size_t)bidx * SEQ + tt) * VV + ncol] = acc[j][r] + bb;
    }
  }
}

// ---- in-place log-softmax over each row of V=32000 (float4 vectorized) -----
__global__ void __launch_bounds__(512)
log_softmax_rows(float* __restrict__ out) {
  float4* row = (float4*)(out + (size_t)blockIdx.x * VV);
  const int NV4 = VV / 4;                       // 8000
  float m = -__builtin_inff(), s = 0.f;
  for (int i = threadIdx.x; i < NV4; i += 512) {
    float4 x = row[i];
    float xm = fmaxf(fmaxf(x.x, x.y), fmaxf(x.z, x.w));
    if (xm > m) { s *= expf(m - xm); m = xm; }
    s += expf(x.x - m) + expf(x.y - m) + expf(x.z - m) + expf(x.w - m);
  }
  __shared__ float sm[512], ss[512];
  sm[threadIdx.x] = m; ss[threadIdx.x] = s;
  __syncthreads();
  for (int k = 256; k >= 1; k >>= 1) {
    if (threadIdx.x < k) {
      float m1 = sm[threadIdx.x], s1 = ss[threadIdx.x];
      float m2 = sm[threadIdx.x + k], s2 = ss[threadIdx.x + k];
      float mm = fmaxf(m1, m2);
      sm[threadIdx.x] = mm;
      ss[threadIdx.x] = s1 * expf(m1 - mm) + s2 * expf(m2 - mm);
    }
    __syncthreads();
  }
  float C = sm[0] + logf(ss[0]);
  for (int i = threadIdx.x; i < NV4; i += 512) {
    float4 x = row[i];
    x.x -= C; x.y -= C; x.z -= C; x.w -= C;
    row[i] = x;
  }
}

// ---------------------------------------------------------------------------
extern "C" void kernel_launch(void* const* d_in, const int* in_sizes, int n_in,
                              void* d_out, int out_size, void* d_ws, size_t ws_size,
                              hipStream_t stream) {
  (void)in_sizes; (void)n_in; (void)out_size; (void)ws_size;
  const int*   src     = (const int*)  d_in[0];
  const int*   tgt     = (const int*)  d_in[1];
  const float* enc_emb = (const float*)d_in[2];
  const float* dec_emb = (const float*)d_in[3];
  const float* Wih_e   = (const float*)d_in[4];
  const float* bih_e   = (const float*)d_in[5];
  const float* Whh_e   = (const float*)d_in[6];
  const float* bhh_e   = (const float*)d_in[7];
  const float* Wih_d   = (const float*)d_in[8];
  const float* bih_d   = (const float*)d_in[9];
  const float* Whh_d   = (const float*)d_in[10];
  const float* bhh_d   = (const float*)d_in[11];
  const float* Wout    = (const float*)d_in[12];
  const float* bout    = (const float*)d_in[13];
  float* out = (float*)d_out;

  // workspace layout (all offsets 256B-aligned)
  char* ws = (char*)d_ws;
  size_t off = 0;
  __bf16* wout_sw  = (__bf16*)(ws + off); off += (size_t)VV * HH * 2;       // 65,536,000
  __bf16* whhe_sw  = (__bf16*)(ws + off); off += (size_t)HH * HH * 2;       //  2,097,152
  __bf16* whhd_sw  = (__bf16*)(ws + off); off += (size_t)HH * HH * 2;
  __bf16* wihe_sw  = (__bf16*)(ws + off); off += (size_t)EE * HH * 2;       //  1,048,576
  __bf16* wihd_sw  = (__bf16*)(ws + off); off += (size_t)EE * HH * 2;
  float*  x_e      = (float*)(ws + off);  off += (size_t)MROWS * HH * 4;    //  8,388,608
  float*  x_d      = (float*)(ws + off);  off += (size_t)MROWS * HH * 4;
  float*  hbuf0    = (float*)(ws + off);  off += (size_t)BB * HH * 4;
  float*  hbuf1    = (float*)(ws + off);  off += (size_t)BB * HH * 4;
  float*  hs       = (float*)(ws + off);  off += (size_t)MROWS * HH * 4;    //  8,388,608
  __bf16* hs_pack  = (__bf16*)(ws + off); off += (size_t)MROWS * HH * 2;    //  4,194,304
  float*  hb[2] = { hbuf0, hbuf1 };

  // 1) weight prep -> bf16 swizzled B fragments
  {
    int n;
    n = (EE / 32) * (HH / 16) * 32;  // 32768
    prep_bsw<<<(n + 255) / 256, 256, 0, stream>>>(Wih_e, wihe_sw, EE, HH);
    prep_bsw<<<(n + 255) / 256, 256, 0, stream>>>(Wih_d, wihd_sw, EE, HH);
    n = (HH / 32) * (HH / 16) * 32;  // 65536
    prep_bsw<<<(n + 255) / 256, 256, 0, stream>>>(Whh_e, whhe_sw, HH, HH);
    prep_bsw<<<(n + 255) / 256, 256, 0, stream>>>(Whh_d, whhd_sw, HH, HH);
    n = (HH / 32) * (VV / 16) * 32;  // 2,048,000
    prep_bsw<<<(n + 255) / 256, 256, 0, stream>>>(Wout, wout_sw, HH, VV);
  }

  // 2) fused embedding gather + input projections
  xproj<<<1024, 256, 0, stream>>>(src, enc_emb, wihe_sw, bih_e, x_e);
  xproj<<<1024, 256, 0, stream>>>(tgt, dec_emb, wihd_sw, bih_d, x_d);

  // 3) h0 = 0
  hipMemsetAsync(hbuf0, 0, (size_t)BB * HH * 4, stream);

  // 4) encoder scan (64 sequential steps)
  for (int t = 0; t < SEQ; ++t)
    rnn_step<<<16, 256, 0, stream>>>(hb[t & 1], x_e, whhe_sw, bhh_e,
                                     hb[(t + 1) & 1], nullptr, t);
  // 5) decoder scan (h_enc lands in hbuf0)
  for (int t = 0; t < SEQ; ++t)
    rnn_step<<<16, 256, 0, stream>>>(hb[t & 1], x_d, whhd_sw, bhh_d,
                                     hb[(t + 1) & 1], hs, t);

  // 6) pack hs into bf16 WMMA A fragments (one pass over 8.4MB)
  {
    int n = (MROWS / 16) * (HH / 32) * 32;  // 131072
    pack_a_bf16<<<(n + 255) / 256, 256, 0, stream>>>(hs, hs_pack,
                                                     MROWS / 16, HH / 32, HH);
  }

  // 7) output projection (writes logits into d_out)
  out_gemm<<<128 * 50, 256, 0, stream>>>(hs_pack, wout_sw, bout, out);

  // 8) in-place log-softmax per (b,t) row
  log_softmax_rows<<<MROWS, 512, 0, stream>>>(out);
}
